// Attention_11424613007451
// MI455X (gfx1250) — matrix operand order repaired and checked by reading, not compiled, not run on hardware
//
#include <hip/hip_runtime.h>

// ---------------------------------------------------------------------------
// MHA forward for MI455X (gfx1250, wave32).
//   x[2,2048,1024] f32, w_qkv[1024,3072] f32, w_out[1024,1024] f32, b_out[1024]
// Pipeline: (1) QKV GEMM -> bf16 q/k/v in [b,h,n,64] (1/32 scale folded into q)
//           (2) flash attention per (b,h), online softmax, bf16 WMMA
//           (3) output GEMM + bias -> f32
// All matmuls: v_wmma_f32_16x16x32_bf16 (f32 accumulate).
// Scheduling: GEMMs software-pipelined; sched_barrier pins next-chunk global
// loads above the WMMA chain; LDS double-buffered (one barrier per chunk);
// B-operands batched -> partial dscnt waits + back-to-back WMMAs.
// bf16 conversion uses the native cast (v_cvt_*bf16_f32, RNE).
// ---------------------------------------------------------------------------

typedef __attribute__((ext_vector_type(16))) __bf16 v16bf;
typedef __attribute__((ext_vector_type(8)))  float  v8f;

struct __attribute__((aligned(16))) U128 { unsigned int w0, w1, w2, w3; };
union V16BF { v16bf v; U128 q[2]; };

__device__ __forceinline__ unsigned short f2bf(float f) {
  __bf16 h = (__bf16)f;                 // native cvt, round-to-nearest-even
  return __builtin_bit_cast(unsigned short, h);
}
__device__ __forceinline__ unsigned int f2bf2(float lo, float hi) {
  return (unsigned int)f2bf(lo) | ((unsigned int)f2bf(hi) << 16);
}

__device__ __forceinline__ v8f wmma_bf16(const V16BF& a, const V16BF& b, v8f c) {
  return __builtin_amdgcn_wmma_f32_16x16x32_bf16(false, a.v, false, b.v,
                                                 (short)0, c, false, false);
}

#define SEQ    2048
#define DMODEL 1024
#define HD     64
#define QSCALE 0.03125f   // 1/sqrt(1024): module scales by full dim

// ---------------------------------------------------------------------------
// Kernel 1: qkv = x @ w_qkv ; scatter to q/k/v [b*h][n][64] bf16, q *= 1/32.
// 256 thr, tile 128x128, K-chunks of 32, waves tiled 4(row)x2(col):
// each wave owns 32 rows x 64 cols -> 2 A-ops, 4 B-ops, 8 WMMAs per chunk.
// grid (24, 32).
// ---------------------------------------------------------------------------
__global__ __launch_bounds__(256) void qkv_gemm(
    const float* __restrict__ x, const float* __restrict__ w,
    unsigned short* __restrict__ qb, unsigned short* __restrict__ kb,
    unsigned short* __restrict__ vb) {
  __shared__ unsigned short Alds[2][128 * 32];   // [m][k] bf16
  __shared__ unsigned short Blds[2][128 * 32];   // [n][k] bf16 (transposed)

  const int tid  = threadIdx.x;
  const int nt   = blockIdx.x;       // 0..23
  const int mt   = blockIdx.y;       // 0..31
  const int wid  = tid >> 5;
  const int lane = tid & 31;
  const int hi   = lane >> 4;
  const int lr   = lane & 15;
  const int rg   = wid >> 1;         // row group: 32 rows
  const int cg   = wid & 1;          // col group: 64 cols

  v8f acc[2][4];
#pragma unroll
  for (int rs = 0; rs < 2; ++rs)
#pragma unroll
    for (int ct = 0; ct < 4; ++ct) acc[rs][ct] = (v8f){0,0,0,0,0,0,0,0};

  // ---- prologue: stage chunk 0 ----
#pragma unroll
  for (int it = 0; it < 4; ++it) {
    int e = (tid + it * 256) * 4;
    int r = e >> 5, c = e & 31;
    float4 f = *(const float4*)(x + (size_t)(mt * 128 + r) * DMODEL + c);
    *(unsigned int*)(&Alds[0][r * 32 + c])     = f2bf2(f.x, f.y);
    *(unsigned int*)(&Alds[0][r * 32 + c + 2]) = f2bf2(f.z, f.w);
  }
#pragma unroll
  for (int it = 0; it < 4; ++it) {
    int e = (tid + it * 256) * 4;
    int r = e >> 7, c = e & 127;
    float4 f = *(const float4*)(w + (size_t)r * 3072 + nt * 128 + c);
    Blds[0][(c + 0) * 32 + r] = f2bf(f.x);
    Blds[0][(c + 1) * 32 + r] = f2bf(f.y);
    Blds[0][(c + 2) * 32 + r] = f2bf(f.z);
    Blds[0][(c + 3) * 32 + r] = f2bf(f.w);
  }
  __syncthreads();

  for (int kc = 0; kc < 32; ++kc) {
    const int cur = kc & 1;
    float4 fa[4], fb[4];
    if (kc + 1 < 32) {                            // global loads for chunk k+1
      const int kk2 = (kc + 1) * 32;
#pragma unroll
      for (int it = 0; it < 4; ++it) {
        int e = (tid + it * 256) * 4;
        int r = e >> 5, c = e & 31;
        fa[it] = *(const float4*)(x + (size_t)(mt * 128 + r) * DMODEL + kk2 + c);
      }
#pragma unroll
      for (int it = 0; it < 4; ++it) {
        int e = (tid + it * 256) * 4;
        int r = e >> 7, c = e & 127;
        fb[it] = *(const float4*)(w + (size_t)(kk2 + r) * 3072 + nt * 128 + c);
      }
    }
    __builtin_amdgcn_sched_barrier(0);            // keep loads above compute

    // ---- compute chunk k from LDS ----
    V16BF a[2], b[4];
#pragma unroll
    for (int rs = 0; rs < 2; ++rs) {
      const unsigned short* ap = &Alds[cur][(rg * 32 + rs * 16 + lr) * 32];
      a[rs].q[0] = *(const U128*)(ap + hi * 8);
      a[rs].q[1] = *(const U128*)(ap + 16 + hi * 8);
    }
#pragma unroll
    for (int ct = 0; ct < 4; ++ct) {
      const unsigned short* bp = &Blds[cur][(cg * 64 + ct * 16 + lr) * 32 + hi * 8];
      b[ct].q[0] = *(const U128*)(bp);
      b[ct].q[1] = *(const U128*)(bp + 16);
    }
#pragma unroll
    for (int rs = 0; rs < 2; ++rs)
#pragma unroll
      for (int ct = 0; ct < 4; ++ct)
        acc[rs][ct] = wmma_bf16(a[rs], b[ct], acc[rs][ct]);

    if (kc + 1 < 32) {                            // stage chunk k+1 (other buf)
      const int nb = cur ^ 1;
#pragma unroll
      for (int it = 0; it < 4; ++it) {
        int e = (tid + it * 256) * 4;
        int r = e >> 5, c = e & 31;
        *(unsigned int*)(&Alds[nb][r * 32 + c])     = f2bf2(fa[it].x, fa[it].y);
        *(unsigned int*)(&Alds[nb][r * 32 + c + 2]) = f2bf2(fa[it].z, fa[it].w);
      }
#pragma unroll
      for (int it = 0; it < 4; ++it) {
        int e = (tid + it * 256) * 4;
        int r = e >> 7, c = e & 127;
        Blds[nb][(c + 0) * 32 + r] = f2bf(fb[it].x);
        Blds[nb][(c + 1) * 32 + r] = f2bf(fb[it].y);
        Blds[nb][(c + 2) * 32 + r] = f2bf(fb[it].z);
        Blds[nb][(c + 3) * 32 + r] = f2bf(fb[it].w);
      }
    }
    __syncthreads();
  }

  // ---- scatter store: C/D layout row = v + 8*hi, col = lr ----
#pragma unroll
  for (int rs = 0; rs < 2; ++rs) {
    const int grb = mt * 128 + rg * 32 + rs * 16 + 8 * hi;
#pragma unroll
    for (int ct = 0; ct < 4; ++ct) {
      int col   = nt * 128 + cg * 64 + ct * 16 + lr;
      int which = col >> 10;
      int cc    = col & 1023;
      int h     = cc >> 6;
      int dd    = cc & 63;
      unsigned short* dst = (which == 0) ? qb : (which == 1) ? kb : vb;
      float mult = (which == 0) ? QSCALE : 1.0f;
#pragma unroll
      for (int v = 0; v < 8; ++v) {
        int gr = grb + v;
        int bb = gr >> 11, nn = gr & 2047;
        dst[((size_t)(bb * 16 + h) * SEQ + nn) * HD + dd] =
            f2bf(acc[rs][ct][v] * mult);
      }
    }
  }
}

// ---------------------------------------------------------------------------
// Kernel 2: flash attention. grid (16 q-tiles, 32 bh). 256 thr.
// Each wave owns 16 query rows x d=64; loop over 64-key tiles with online
// softmax. K/V global loads for tile j+1 overlap the compute of tile j.
// LDS: Q 16K, K 8K, V^T 8K, P 16K = 48 KB.
// ---------------------------------------------------------------------------
__global__ __launch_bounds__(256) void flash_attn(
    const unsigned short* __restrict__ qb, const unsigned short* __restrict__ kb,
    const unsigned short* __restrict__ vb, unsigned short* __restrict__ ob) {
  __shared__ unsigned short Qlds[128 * 64];
  __shared__ unsigned short Klds[64 * 64];
  __shared__ unsigned short Vt[64 * 64];          // [d][key]
  __shared__ unsigned short Plds[8 * 16 * 64];    // per-wave 16x64 strip

  const int tid  = threadIdx.x;
  const int wid  = tid >> 5;
  const int lane = tid & 31;
  const int hi   = lane >> 4;
  const int lr   = lane & 15;
  const int qt   = blockIdx.x;
  const int bh   = blockIdx.y;
  const size_t base = (size_t)bh * SEQ * HD;

#pragma unroll
  for (int it = 0; it < 4; ++it) {                // Q tile 128x64
    int e = (tid + it * 256) * 8;
    int r = e >> 6, c = e & 63;
    *(U128*)(&Qlds[r * 64 + c]) =
        *(const U128*)(&qb[base + (size_t)(qt * 128 + r) * HD + c]);
  }
  {                                               // K/V tile 0
#pragma unroll
    for (int it = 0; it < 2; ++it) {
      int e = (tid + it * 256) * 8;
      int r = e >> 6, c = e & 63;
      *(U128*)(&Klds[r * 64 + c]) = *(const U128*)(&kb[base + (size_t)r * HD + c]);
      union { U128 u; unsigned short s[8]; } t;
      t.u = *(const U128*)(&vb[base + (size_t)r * HD + c]);
#pragma unroll
      for (int i = 0; i < 8; ++i) Vt[(c + i) * 64 + r] = t.s[i];
    }
  }
  __syncthreads();

  V16BF aq[2];                                    // Q A-operands: registers
#pragma unroll
  for (int kq = 0; kq < 2; ++kq) {
    const unsigned short* qp = &Qlds[(wid * 16 + lr) * 64 + kq * 32];
    aq[kq].q[0] = *(const U128*)(qp + hi * 8);
    aq[kq].q[1] = *(const U128*)(qp + 16 + hi * 8);
  }

  float m_i[8], l_i[8];
  v8f O[4];
#pragma unroll
  for (int v = 0; v < 8; ++v) { m_i[v] = -1.0e30f; l_i[v] = 0.0f; }
#pragma unroll
  for (int dt = 0; dt < 4; ++dt) O[dt] = (v8f){0,0,0,0,0,0,0,0};

  for (int jt = 0; jt < SEQ / 64; ++jt) {
    U128 uk[2], uv[2];
    if (jt + 1 < SEQ / 64) {                      // prefetch tile j+1 into regs
#pragma unroll
      for (int it = 0; it < 2; ++it) {
        int e = (tid + it * 256) * 8;
        int r = e >> 6, c = e & 63;
        uk[it] = *(const U128*)(&kb[base + (size_t)((jt + 1) * 64 + r) * HD + c]);
        uv[it] = *(const U128*)(&vb[base + (size_t)((jt + 1) * 64 + r) * HD + c]);
      }
    }
    __builtin_amdgcn_sched_barrier(0);            // keep loads above compute

    v8f S[4];                                     // S = Q K^T  (16 x 64)
#pragma unroll
    for (int ct = 0; ct < 4; ++ct) S[ct] = (v8f){0,0,0,0,0,0,0,0};
#pragma unroll
    for (int kq = 0; kq < 2; ++kq) {
      V16BF bk[4];
#pragma unroll
      for (int ct = 0; ct < 4; ++ct) {
        const unsigned short* bp = &Klds[(ct * 16 + lr) * 64 + kq * 32 + hi * 8];
        bk[ct].q[0] = *(const U128*)bp;
        bk[ct].q[1] = *(const U128*)(bp + 16);
      }
#pragma unroll
      for (int ct = 0; ct < 4; ++ct) S[ct] = wmma_bf16(aq[kq], bk[ct], S[ct]);
    }

#pragma unroll
    for (int v = 0; v < 8; ++v) {                 // online softmax per row
      float mx = S[0][v];
#pragma unroll
      for (int ct = 1; ct < 4; ++ct) mx = fmaxf(mx, S[ct][v]);
#pragma unroll
      for (int off = 1; off < 16; off <<= 1) mx = fmaxf(mx, __shfl_xor(mx, off, 32));
      float m_new = fmaxf(m_i[v], mx);
      float corr  = __expf(m_i[v] - m_new);
      m_i[v] = m_new;
      float rs = 0.0f;
#pragma unroll
      for (int ct = 0; ct < 4; ++ct) {
        float p = __expf(S[ct][v] - m_new);
        rs += p;
        Plds[wid * 1024 + (v + 8 * hi) * 64 + ct * 16 + lr] = f2bf(p);
      }
#pragma unroll
      for (int off = 1; off < 16; off <<= 1) rs += __shfl_xor(rs, off, 32);
      l_i[v] = l_i[v] * corr + rs;
#pragma unroll
      for (int dt = 0; dt < 4; ++dt) O[dt][v] *= corr;
    }

#pragma unroll
    for (int kc = 0; kc < 2; ++kc) {              // O += P V
      V16BF ap, bv[4];
      const unsigned short* pp = &Plds[wid * 1024 + lr * 64 + kc * 32 + hi * 8];
      ap.q[0] = *(const U128*)pp;
      ap.q[1] = *(const U128*)(pp + 16);
#pragma unroll
      for (int dt = 0; dt < 4; ++dt) {
        const unsigned short* vp = &Vt[(dt * 16 + lr) * 64 + kc * 32 + hi * 8];
        bv[dt].q[0] = *(const U128*)vp;
        bv[dt].q[1] = *(const U128*)(vp + 16);
      }
#pragma unroll
      for (int dt = 0; dt < 4; ++dt) O[dt] = wmma_bf16(ap, bv[dt], O[dt]);
    }

    __syncthreads();                              // all waves done with K/Vt
    if (jt + 1 < SEQ / 64) {                      // stage tile j+1
#pragma unroll
      for (int it = 0; it < 2; ++it) {
        int e = (tid + it * 256) * 8;
        int r = e >> 6, c = e & 63;
        *(U128*)(&Klds[r * 64 + c]) = uk[it];
        union { U128 u; unsigned short s[8]; } t;
        t.u = uv[it];
#pragma unroll
        for (int i = 0; i < 8; ++i) Vt[(c + i) * 64 + r] = t.s[i];
      }
    }
    __syncthreads();
  }

  const int bb = bh >> 4, h = bh & 15;
#pragma unroll
  for (int v = 0; v < 8; ++v) {
    float inv = 1.0f / l_i[v];
    int nn = qt * 128 + wid * 16 + v + 8 * hi;
#pragma unroll
    for (int dt = 0; dt < 4; ++dt)
      ob[((size_t)(bb * SEQ + nn)) * DMODEL + h * HD + dt * 16 + lr] =
          f2bf(O[dt][v] * inv);
  }
}

// ---------------------------------------------------------------------------
// Kernel 3: out = o @ w_out + b_out (f32 out). grid (8, 32). Same pipelined
// structure as kernel 1; A is already bf16.
// ---------------------------------------------------------------------------
__global__ __launch_bounds__(256) void out_gemm(
    const unsigned short* __restrict__ ob, const float* __restrict__ w,
    const float* __restrict__ bias, float* __restrict__ out) {
  __shared__ unsigned short Alds[2][128 * 32];
  __shared__ unsigned short Blds[2][128 * 32];

  const int tid  = threadIdx.x;
  const int nt   = blockIdx.x;      // 0..7
  const int mt   = blockIdx.y;      // 0..31
  const int wid  = tid >> 5;
  const int lane = tid & 31;
  const int hi   = lane >> 4;
  const int lr   = lane & 15;
  const int rg   = wid >> 1;
  const int cg   = wid & 1;

  v8f acc[2][4];
#pragma unroll
  for (int rs = 0; rs < 2; ++rs)
#pragma unroll
    for (int ct = 0; ct < 4; ++ct) acc[rs][ct] = (v8f){0,0,0,0,0,0,0,0};

#pragma unroll
  for (int it = 0; it < 2; ++it) {                // chunk 0: A (bf16 copy)
    int e = (tid + it * 256) * 8;
    int r = e >> 5, c = e & 31;
    *(U128*)(&Alds[0][r * 32 + c]) =
        *(const U128*)(&ob[(size_t)(mt * 128 + r) * DMODEL + c]);
  }
#pragma unroll
  for (int it = 0; it < 4; ++it) {                // chunk 0: B (f32 -> bf16, T)
    int e = (tid + it * 256) * 4;
    int r = e >> 7, c = e & 127;
    float4 f = *(const float4*)(w + (size_t)r * DMODEL + nt * 128 + c);
    Blds[0][(c + 0) * 32 + r] = f2bf(f.x);
    Blds[0][(c + 1) * 32 + r] = f2bf(f.y);
    Blds[0][(c + 2) * 32 + r] = f2bf(f.z);
    Blds[0][(c + 3) * 32 + r] = f2bf(f.w);
  }
  __syncthreads();

  for (int kc = 0; kc < 32; ++kc) {
    const int cur = kc & 1;
    U128 ua[2];
    float4 fb[4];
    if (kc + 1 < 32) {
      const int kk2 = (kc + 1) * 32;
#pragma unroll
      for (int it = 0; it < 2; ++it) {
        int e = (tid + it * 256) * 8;
        int r = e >> 5, c = e & 31;
        ua[it] = *(const U128*)(&ob[(size_t)(mt * 128 + r) * DMODEL + kk2 + c]);
      }
#pragma unroll
      for (int it = 0; it < 4; ++it) {
        int e = (tid + it * 256) * 4;
        int r = e >> 7, c = e & 127;
        fb[it] = *(const float4*)(w + (size_t)(kk2 + r) * DMODEL + nt * 128 + c);
      }
    }
    __builtin_amdgcn_sched_barrier(0);            // keep loads above compute

    V16BF a[2], b[4];
#pragma unroll
    for (int rs = 0; rs < 2; ++rs) {
      const unsigned short* ap = &Alds[cur][(rg * 32 + rs * 16 + lr) * 32];
      a[rs].q[0] = *(const U128*)(ap + hi * 8);
      a[rs].q[1] = *(const U128*)(ap + 16 + hi * 8);
    }
#pragma unroll
    for (int ct = 0; ct < 4; ++ct) {
      const unsigned short* bp = &Blds[cur][(cg * 64 + ct * 16 + lr) * 32 + hi * 8];
      b[ct].q[0] = *(const U128*)(bp);
      b[ct].q[1] = *(const U128*)(bp + 16);
    }
#pragma unroll
    for (int rs = 0; rs < 2; ++rs)
#pragma unroll
      for (int ct = 0; ct < 4; ++ct)
        acc[rs][ct] = wmma_bf16(a[rs], b[ct], acc[rs][ct]);

    if (kc + 1 < 32) {
      const int nb = cur ^ 1;
#pragma unroll
      for (int it = 0; it < 2; ++it) {
        int e = (tid + it * 256) * 8;
        int r = e >> 5, c = e & 31;
        *(U128*)(&Alds[nb][r * 32 + c]) = ua[it];
      }
#pragma unroll
      for (int it = 0; it < 4; ++it) {
        int e = (tid + it * 256) * 4;
        int r = e >> 7, c = e & 127;
        Blds[nb][(c + 0) * 32 + r] = f2bf(fb[it].x);
        Blds[nb][(c + 1) * 32 + r] = f2bf(fb[it].y);
        Blds[nb][(c + 2) * 32 + r] = f2bf(fb[it].z);
        Blds[nb][(c + 3) * 32 + r] = f2bf(fb[it].w);
      }
    }
    __syncthreads();
  }

#pragma unroll
  for (int rs = 0; rs < 2; ++rs) {
    const int grb = mt * 128 + rg * 32 + rs * 16 + 8 * hi;
#pragma unroll
    for (int ct = 0; ct < 4; ++ct) {
      int col  = nt * 128 + cg * 64 + ct * 16 + lr;
      float bo = bias[col];
#pragma unroll
      for (int v = 0; v < 8; ++v)
        out[(size_t)(grb + v) * DMODEL + col] = acc[rs][ct][v] + bo;
    }
  }
}

// ---------------------------------------------------------------------------
extern "C" void kernel_launch(void* const* d_in, const int* in_sizes, int n_in,
                              void* d_out, int out_size, void* d_ws, size_t ws_size,
                              hipStream_t stream) {
  const float* x     = (const float*)d_in[0];   // [2,2048,1024]
  const float* w_qkv = (const float*)d_in[1];   // [1024,3072]
  const float* w_out = (const float*)d_in[2];   // [1024,1024]
  const float* b_out = (const float*)d_in[3];   // [1024]
  float* out = (float*)d_out;                   // [2,2048,1024]

  const size_t QKV_ELEMS = (size_t)2 * 16 * SEQ * HD;  // 4,194,304 per tensor
  unsigned short* qb = (unsigned short*)d_ws;
  unsigned short* kb = qb + QKV_ELEMS;
  unsigned short* vb = kb + QKV_ELEMS;
  unsigned short* ob = vb + QKV_ELEMS;                 // [b,n,1024] bf16

  qkv_gemm<<<dim3(24, 32), 256, 0, stream>>>(x, w_qkv, qb, kb, vb);
  flash_attn<<<dim3(16, 32), 256, 0, stream>>>(qb, kb, vb, ob);
  out_gemm<<<dim3(8, 32), 256, 0, stream>>>(ob, w_out, b_out, out);
}
